// LLM_26869315404030
// MI455X (gfx1250) — matrix operand order repaired and checked by reading, not compiled, not run on hardware
//
#include <hip/hip_runtime.h>
#include <hip/hip_bf16.h>
#include <math.h>
#include <stdint.h>

// ---------------- model constants (match reference) ----------------
#define D_MODEL   1024
#define N_HEAD    16
#define HEAD_DIM  64
#define N_LAYER   8
#define BATCH     2
#define SEQ       1024
#define ROWS      (BATCH * SEQ)   // 2048
#define VOCAB     50257

typedef _Float16 half_t;
typedef __attribute__((ext_vector_type(16))) _Float16 v16h;
typedef __attribute__((ext_vector_type(8)))  float    v8f;

// ---------------- embedding: x = tok_emb[idx] + pos_emb[t] ----------------
__global__ void embed_kernel(const int* __restrict__ idx,
                             const float* __restrict__ tok,
                             const float* __restrict__ pos,
                             float* __restrict__ x) {
    int row = blockIdx.x;            // b*SEQ + t
    int t   = row % SEQ;
    int token = idx[row];
    const float* te = tok + (size_t)token * D_MODEL;
    const float* pe = pos + (size_t)t * D_MODEL;
    float* xr = x + (size_t)row * D_MODEL;
    for (int c = threadIdx.x; c < D_MODEL; c += blockDim.x)
        xr[c] = te[c] + pe[c];
}

// ---------------- layernorm (f32 in, f16 out for WMMA A-operand) ----------------
__global__ __launch_bounds__(256)
void ln_kernel(const float* __restrict__ x,
               const float* __restrict__ g,
               const float* __restrict__ b,
               half_t* __restrict__ out16) {
    __shared__ float red[256];
    int row = blockIdx.x;
    int tid = threadIdx.x;
    const float* xr = x + (size_t)row * D_MODEL;

    float s = 0.f;
    for (int c = tid; c < D_MODEL; c += 256) s += xr[c];
    red[tid] = s; __syncthreads();
    for (int o = 128; o > 0; o >>= 1) {
        if (tid < o) red[tid] += red[tid + o];
        __syncthreads();
    }
    float mean = red[0] * (1.0f / D_MODEL);
    __syncthreads();

    float v = 0.f;
    for (int c = tid; c < D_MODEL; c += 256) {
        float d = xr[c] - mean; v += d * d;
    }
    red[tid] = v; __syncthreads();
    for (int o = 128; o > 0; o >>= 1) {
        if (tid < o) red[tid] += red[tid + o];
        __syncthreads();
    }
    float rstd = rsqrtf(red[0] * (1.0f / D_MODEL) + 1e-5f);

    half_t* orow = out16 + (size_t)row * D_MODEL;
    for (int c = tid; c < D_MODEL; c += 256)
        orow[c] = (half_t)((xr[c] - mean) * rstd * g[c] + b[c]);
}

// ---------------- weight transpose + convert: W[K,N] f32 -> Wt[N,K] f16 ----------------
__global__ __launch_bounds__(256)
void transpose_f16_kernel(const float* __restrict__ in, half_t* __restrict__ out,
                          int K, int N) {
    __shared__ float tile[32][33];
    int kb = blockIdx.y * 32, nb = blockIdx.x * 32;
    int tx = threadIdx.x & 31, ty = threadIdx.x >> 5;   // 8 rows per pass
#pragma unroll
    for (int i = 0; i < 32; i += 8)
        tile[ty + i][tx] = in[(size_t)(kb + ty + i) * N + (nb + tx)];
    __syncthreads();
#pragma unroll
    for (int i = 0; i < 32; i += 8)
        out[(size_t)(nb + ty + i) * K + (kb + tx)] = (half_t)tile[tx][ty + i];
}

// ---------------- elementwise f32 -> f16 (tok_emb is already [V,D]=[N,K]) ----------------
__global__ void f32_to_f16_kernel(const float* __restrict__ in,
                                  half_t* __restrict__ out, size_t n) {
    size_t i = (size_t)blockIdx.x * blockDim.x + threadIdx.x;
    size_t stride = (size_t)gridDim.x * blockDim.x;
    for (; i < n; i += stride) out[i] = (half_t)in[i];
}

// ---------------- shared WMMA tile compute ----------------
// pA: LDS [128][40] f16 tile [m][k]; pB: LDS [128][40] f16 tile [n][k].
// Each wave computes a 32x64 region = 2x4 v_wmma_f32_16x16x32_f16.
__device__ __forceinline__
void wmma_tile_compute(const half_t (*pA)[40], const half_t (*pB)[40],
                       int lane, int wm, int wn, v8f acc[2][4]) {
    v16h afrag[2];
#pragma unroll
    for (int ti = 0; ti < 2; ti++) {
        int m  = wm + ti * 16 + (lane & 15);
        int kh = (lane < 16) ? 0 : 8;
#pragma unroll
        for (int i = 0; i < 8; i++) {
            int ko = ((i < 4) ? 0 : 16) + kh + 2 * (i & 3);   // ISA 16-bit A 16x32 layout
            afrag[ti][2 * i]     = pA[m][ko];
            afrag[ti][2 * i + 1] = pA[m][ko + 1];
        }
    }
#pragma unroll
    for (int tj = 0; tj < 4; tj++) {
        int n  = wn + tj * 16 + (lane & 15);
        int kb = (lane < 16) ? 0 : 8;                         // B: same K layout as A
        v16h bfrag;
#pragma unroll
        for (int i = 0; i < 8; i++) {
            int ko = ((i < 4) ? 0 : 16) + kb + 2 * (i & 3);
            bfrag[2 * i]     = pB[n][ko];
            bfrag[2 * i + 1] = pB[n][ko + 1];
        }
#pragma unroll
        for (int ti = 0; ti < 2; ti++) {
            acc[ti][tj] = __builtin_amdgcn_wmma_f32_16x16x32_f16(
                false, afrag[ti], false, bfrag, (short)0, acc[ti][tj], false, false);
        }
    }
}

// ---------------- WMMA GEMM, all-f16 operands, async double-buffered staging ----------------
// C[M,N] = epi(A[M,K] @ Bt[N,K]^T). Per-lane global_load_async_to_lds_b128
// (ASYNCcnt) fills the next LDS buffer while the current feeds WMMA. All
// staging addresses are hoisted: 4 global pointers advance by 32 halves per
// k-step; LDS offsets are fixed per buffer. LDS pitch 40 halves (80B):
// 16B-aligned b128 chunks, conflict-free (20*m mod 64) bank pattern.
template <bool GELU_ACT>
__global__ __launch_bounds__(256)
void gemm_f16_kernel(const half_t* __restrict__ A,    // [M,K] f16
                     const half_t* __restrict__ Bt,   // [N,K] f16
                     const float* __restrict__ bias,  // [N] or null
                     const float* __restrict__ resid, // [M,N] or null
                     float* __restrict__ C,           // [M,N] or null
                     half_t* __restrict__ C16,        // [M,N] or null
                     int M, int N, int K) {
    __shared__ __align__(16) half_t sA[2][128][40];
    __shared__ __align__(16) half_t sB[2][128][40];

    const int tid  = threadIdx.x;
    const int lane = tid & 31;
    const int wave = tid >> 5;
    const int wm   = (wave >> 1) * 32;
    const int wn   = (wave & 1) * 64;
    const int m0   = blockIdx.y * 128;
    const int n0   = blockIdx.x * 128;

    const uint32_t ldsA0 = (uint32_t)(uintptr_t)&sA[0][0][0];
    const uint32_t ldsB0 = (uint32_t)(uintptr_t)&sB[0][0][0];
    const uint32_t BUFB  = 128u * 40u * 2u;   // 10240 bytes per buffer

    // ---- hoisted per-thread staging descriptors: 2 chunks of A, 2 of B ----
    const int q0 = tid, q1 = tid + 256;       // 512 16B-chunks per 128x32 tile
    const int r0 = q0 >> 2, c0 = q0 & 3;      // row 0..127, chunk 0..3
    const int r1 = q1 >> 2, c1 = q1 & 3;
    int gn0 = n0 + r0; if (gn0 > N - 1) gn0 = N - 1;   // clamp at N edge
    int gn1 = n0 + r1; if (gn1 > N - 1) gn1 = N - 1;   // (masked in epilogue)
    const half_t* gA0 = A  + (size_t)(m0 + r0) * K + c0 * 8;
    const half_t* gA1 = A  + (size_t)(m0 + r1) * K + c1 * 8;
    const half_t* gB0 = Bt + (size_t)gn0 * K + c0 * 8;
    const half_t* gB1 = Bt + (size_t)gn1 * K + c1 * 8;
    const uint32_t laA0 = ldsA0 + (uint32_t)(r0 * 80 + c0 * 16);
    const uint32_t laA1 = ldsA0 + (uint32_t)(r1 * 80 + c1 * 16);
    const uint32_t laB0 = ldsB0 + (uint32_t)(r0 * 80 + c0 * 16);
    const uint32_t laB1 = ldsB0 + (uint32_t)(r1 * 80 + c1 * 16);

    v8f acc[2][4];
#pragma unroll
    for (int i = 0; i < 2; i++)
#pragma unroll
        for (int j = 0; j < 4; j++)
#pragma unroll
            for (int e = 0; e < 8; e++) acc[i][j][e] = 0.f;

    const int nsteps = K >> 5;

    auto stage = [&](uint32_t bufoff) {
        asm volatile("global_load_async_to_lds_b128 %0, %1, off"
                     :: "v"(laA0 + bufoff), "v"((uint64_t)(uintptr_t)gA0) : "memory");
        asm volatile("global_load_async_to_lds_b128 %0, %1, off"
                     :: "v"(laA1 + bufoff), "v"((uint64_t)(uintptr_t)gA1) : "memory");
        asm volatile("global_load_async_to_lds_b128 %0, %1, off"
                     :: "v"(laB0 + bufoff), "v"((uint64_t)(uintptr_t)gB0) : "memory");
        asm volatile("global_load_async_to_lds_b128 %0, %1, off"
                     :: "v"(laB1 + bufoff), "v"((uint64_t)(uintptr_t)gB1) : "memory");
        gA0 += 32; gA1 += 32; gB0 += 32; gB1 += 32;   // next k-tile (+64B)
    };

    stage(0);
    for (int s = 0; s < nsteps; s++) {
        asm volatile("s_wait_asynccnt 0x0" ::: "memory");   // this wave's copies done
        __syncthreads();                                    // all waves' copies visible
        if (s + 1 < nsteps) stage((uint32_t)((s + 1) & 1) * BUFB);
        wmma_tile_compute(sA[s & 1], sB[s & 1], lane, wm, wn, acc);
        __syncthreads();                                    // reads drained before overwrite
    }

    // ---- epilogue + store (ISA 32-bit C/D 16x16 layout) ----
#pragma unroll
    for (int ti = 0; ti < 2; ti++) {
#pragma unroll
        for (int tj = 0; tj < 4; tj++) {
#pragma unroll
            for (int v = 0; v < 8; v++) {
                int row = m0 + wm + ti * 16 + v + ((lane >= 16) ? 8 : 0);
                int col = n0 + wn + tj * 16 + (lane & 15);
                if (col < N) {
                    float val = acc[ti][tj][v];
                    if (bias) val += bias[col];
                    if (GELU_ACT)
                        val = 0.5f * val * (1.f + erff(val * 0.70710678118654752f));
                    if (resid) val += resid[(size_t)row * N + col];
                    if (C)   C  [(size_t)row * N + col] = val;
                    if (C16) C16[(size_t)row * N + col] = (half_t)val;
                }
            }
        }
    }
}

// ---------------- fallback logits GEMM (B = f32 [N,K], converted on the fly) ----------------
__global__ __launch_bounds__(256)
void gemm_logits_f32b_kernel(const half_t* __restrict__ A,  // [M,K] f16
                             const float* __restrict__ Bg,  // [N,K] f32 (tok_emb)
                             float* __restrict__ C,         // [M,N]
                             int M, int N, int K) {
    __shared__ half_t sA[128][40];
    __shared__ half_t sB[128][40];
    const int tid  = threadIdx.x;
    const int lane = tid & 31;
    const int wave = tid >> 5;
    const int wm   = (wave >> 1) * 32;
    const int wn   = (wave & 1) * 64;
    const int m0   = blockIdx.y * 128;
    const int n0   = blockIdx.x * 128;

    v8f acc[2][4];
#pragma unroll
    for (int i = 0; i < 2; i++)
#pragma unroll
        for (int j = 0; j < 4; j++)
#pragma unroll
            for (int e = 0; e < 8; e++) acc[i][j][e] = 0.f;

    for (int k0 = 0; k0 < K; k0 += 32) {
        for (int i = tid; i < 128 * 32; i += 256) {
            int r = i >> 5, c = i & 31;
            sA[r][c] = A[(size_t)(m0 + r) * K + k0 + c];
        }
        for (int i = tid; i < 128 * 32; i += 256) {
            int r = i >> 5, c = i & 31;
            int gn = n0 + r;
            sB[r][c] = (half_t)((gn < N) ? Bg[(size_t)gn * K + k0 + c] : 0.f);
        }
        __syncthreads();
        wmma_tile_compute(sA, sB, lane, wm, wn, acc);
        __syncthreads();
    }
#pragma unroll
    for (int ti = 0; ti < 2; ti++)
#pragma unroll
        for (int tj = 0; tj < 4; tj++)
#pragma unroll
            for (int v = 0; v < 8; v++) {
                int row = m0 + wm + ti * 16 + v + ((lane >= 16) ? 8 : 0);
                int col = n0 + wn + tj * 16 + (lane & 15);
                if (col < N) C[(size_t)row * N + col] = acc[ti][tj][v];
            }
}

// ---------------- attention: streaming online-softmax, f16 output ----------------
__global__ __launch_bounds__(256)
void attn_kernel(const float* __restrict__ qkv, half_t* __restrict__ out16) {
    int gid = blockIdx.x * blockDim.x + threadIdx.x;
    if (gid >= ROWS * N_HEAD) return;
    int h   = gid % N_HEAD;
    int row = gid / N_HEAD;      // b*SEQ + t
    int t   = row % SEQ;
    int b   = row / SEQ;

    const float* qrow = qkv + (size_t)row * (3 * D_MODEL) + h * HEAD_DIM;
    float q[HEAD_DIM], o[HEAD_DIM];
#pragma unroll
    for (int d = 0; d < HEAD_DIM; d++) { q[d] = qrow[d] * 0.125f; o[d] = 0.f; }

    const float* kbase = qkv + (size_t)(b * SEQ) * (3 * D_MODEL) + D_MODEL     + h * HEAD_DIM;
    const float* vbase = qkv + (size_t)(b * SEQ) * (3 * D_MODEL) + 2 * D_MODEL + h * HEAD_DIM;

    float mmax = -INFINITY, denom = 0.f;
    for (int kk = 0; kk <= t; kk++) {
        const float* kr = kbase + (size_t)kk * (3 * D_MODEL);
        float s = 0.f;
#pragma unroll
        for (int d = 0; d < HEAD_DIM; d++) s += q[d] * kr[d];
        float mnew = fmaxf(mmax, s);
        float corr = __expf(mmax - mnew);
        float p    = __expf(s - mnew);
        denom = denom * corr + p;
        const float* vr = vbase + (size_t)kk * (3 * D_MODEL);
#pragma unroll
        for (int d = 0; d < HEAD_DIM; d++) o[d] = o[d] * corr + p * vr[d];
        mmax = mnew;
    }
    float inv = 1.f / denom;
    half_t* orow = out16 + (size_t)row * D_MODEL + h * HEAD_DIM;
#pragma unroll
    for (int d = 0; d < HEAD_DIM; d++) orow[d] = (half_t)(o[d] * inv);
}

// ---------------- loss (deterministic: per-row partials + fixed-order reduce) ----------------
__global__ __launch_bounds__(256)
void loss_row_kernel(const float* __restrict__ logits,
                     const int* __restrict__ targets,
                     float* __restrict__ rowloss) {
    __shared__ float red[256];
    int row = blockIdx.x;
    int tid = threadIdx.x;
    const float* lr = logits + (size_t)row * VOCAB;

    float m = -INFINITY;
    for (int c = tid; c < VOCAB; c += 256) m = fmaxf(m, lr[c]);
    red[tid] = m; __syncthreads();
    for (int o = 128; o > 0; o >>= 1) {
        if (tid < o) red[tid] = fmaxf(red[tid], red[tid + o]);
        __syncthreads();
    }
    m = red[0]; __syncthreads();

    float s = 0.f;
    for (int c = tid; c < VOCAB; c += 256) s += __expf(lr[c] - m);
    red[tid] = s; __syncthreads();
    for (int o = 128; o > 0; o >>= 1) {
        if (tid < o) red[tid] += red[tid + o];
        __syncthreads();
    }
    if (tid == 0)
        rowloss[row] = (m + __logf(red[0])) - lr[targets[row]];
}

__global__ __launch_bounds__(256)
void loss_reduce_kernel(const float* __restrict__ rowloss, float* __restrict__ loss) {
    __shared__ float red[256];
    float s = 0.f;
    for (int i = threadIdx.x; i < ROWS; i += 256) s += rowloss[i];
    red[threadIdx.x] = s; __syncthreads();
    for (int o = 128; o > 0; o >>= 1) {
        if (threadIdx.x < o) red[threadIdx.x] += red[threadIdx.x + o];
        __syncthreads();
    }
    if (threadIdx.x == 0) *loss = red[0] * (1.0f / ROWS);
}

// ---------------- driver ----------------
extern "C" void kernel_launch(void* const* d_in, const int* in_sizes, int n_in,
                              void* d_out, int out_size, void* d_ws, size_t ws_size,
                              hipStream_t stream) {
    const int*   idx     = (const int*)  d_in[0];
    const int*   targets = (const int*)  d_in[1];
    const float* tok     = (const float*)d_in[2];
    const float* pos     = (const float*)d_in[3];
    const float* ln1_g   = (const float*)d_in[4];
    const float* ln1_b   = (const float*)d_in[5];
    const float* qkv_w   = (const float*)d_in[6];
    const float* proj_w  = (const float*)d_in[7];
    const float* ln2_g   = (const float*)d_in[8];
    const float* ln2_b   = (const float*)d_in[9];
    const float* w1      = (const float*)d_in[10];
    const float* b1      = (const float*)d_in[11];
    const float* w2      = (const float*)d_in[12];
    const float* b2      = (const float*)d_in[13];
    const float* lnf_g   = (const float*)d_in[14];
    const float* lnf_b   = (const float*)d_in[15];

    float* out = (float*)d_out;                 // [ROWS*VOCAB logits][1 loss]
    char*  ws  = (char*)d_ws;

    // workspace layout (bytes)
    const size_t X_OFF    = 0;                                             // f32 [ROWS,D]   8 MB
    const size_t QKV_OFF  = X_OFF   + (size_t)ROWS * D_MODEL * 4;          // f32 [ROWS,3D] 24 MB
    const size_t HH_OFF   = QKV_OFF + (size_t)ROWS * 3 * D_MODEL * 4;      // f16 [ROWS,D]   4 MB
    const size_t ATT_OFF  = HH_OFF  + (size_t)ROWS * D_MODEL * 2;          // f16 [ROWS,D]   4 MB
    const size_t ACT_OFF  = ATT_OFF + (size_t)ROWS * D_MODEL * 2;          // f16 [ROWS,4D] 16 MB
    const size_t WT_OFF   = ACT_OFF + (size_t)ROWS * 4 * D_MODEL * 2;      // f16 weight scratch
    const size_t WT_SZ    = (size_t)4 * D_MODEL * D_MODEL * 2;             // 8 MB (max weight)
    const size_t TOK_OFF  = WT_OFF + WT_SZ;                                // f16 [V,D]  ~98 MB
    const size_t TOK_SZ   = (size_t)VOCAB * D_MODEL * 2;

    float*  x      = (float*) (ws + X_OFF);
    float*  qkvb   = (float*) (ws + QKV_OFF);
    half_t* hh     = (half_t*)(ws + HH_OFF);
    half_t* attn16 = (half_t*)(ws + ATT_OFF);
    half_t* act16  = (half_t*)(ws + ACT_OFF);
    half_t* wt     = (half_t*)(ws + WT_OFF);
    half_t* tok16  = (half_t*)(ws + TOK_OFF);
    float*  rowls  = (float*) (ws + WT_OFF);    // reuse weight scratch at loss time

    const bool use_tok16 = (ws_size >= TOK_OFF + TOK_SZ);

    dim3 blk(256);

    if (use_tok16)  // f16 copy of tied embedding for the logits GEMM B-operand
        f32_to_f16_kernel<<<2048, blk, 0, stream>>>(tok, tok16, (size_t)VOCAB * D_MODEL);

    embed_kernel<<<ROWS, blk, 0, stream>>>(idx, tok, pos, x);

    for (int l = 0; l < N_LAYER; l++) {
        const float* Wqkv = qkv_w  + (size_t)l * D_MODEL * 3 * D_MODEL;
        const float* Wp   = proj_w + (size_t)l * D_MODEL * D_MODEL;
        const float* W1l  = w1     + (size_t)l * D_MODEL * 4 * D_MODEL;
        const float* B1l  = b1     + (size_t)l * 4 * D_MODEL;
        const float* W2l  = w2     + (size_t)l * 4 * D_MODEL * D_MODEL;
        const float* B2l  = b2     + (size_t)l * D_MODEL;

        // h = LN1(x)  (f16)
        ln_kernel<<<ROWS, blk, 0, stream>>>(x, ln1_g + l * D_MODEL, ln1_b + l * D_MODEL, hh);
        // qkv = h @ Wqkv
        transpose_f16_kernel<<<dim3(3 * D_MODEL / 32, D_MODEL / 32), blk, 0, stream>>>(
            Wqkv, wt, D_MODEL, 3 * D_MODEL);
        gemm_f16_kernel<false><<<dim3(3 * D_MODEL / 128, ROWS / 128), blk, 0, stream>>>(
            hh, wt, nullptr, nullptr, qkvb, nullptr, ROWS, 3 * D_MODEL, D_MODEL);
        // attention (f16 out)
        attn_kernel<<<(ROWS * N_HEAD + 255) / 256, blk, 0, stream>>>(qkvb, attn16);
        // x = x + attn @ Wp
        transpose_f16_kernel<<<dim3(D_MODEL / 32, D_MODEL / 32), blk, 0, stream>>>(
            Wp, wt, D_MODEL, D_MODEL);
        gemm_f16_kernel<false><<<dim3(D_MODEL / 128, ROWS / 128), blk, 0, stream>>>(
            attn16, wt, nullptr, x, x, nullptr, ROWS, D_MODEL, D_MODEL);
        // h = LN2(x)  (f16)
        ln_kernel<<<ROWS, blk, 0, stream>>>(x, ln2_g + l * D_MODEL, ln2_b + l * D_MODEL, hh);
        // act = gelu(h @ W1 + b1)  (f16 only — consumed as next A-operand)
        transpose_f16_kernel<<<dim3(4 * D_MODEL / 32, D_MODEL / 32), blk, 0, stream>>>(
            W1l, wt, D_MODEL, 4 * D_MODEL);
        gemm_f16_kernel<true><<<dim3(4 * D_MODEL / 128, ROWS / 128), blk, 0, stream>>>(
            hh, wt, B1l, nullptr, nullptr, act16, ROWS, 4 * D_MODEL, D_MODEL);
        // x = x + act @ W2 + b2
        transpose_f16_kernel<<<dim3(D_MODEL / 32, 4 * D_MODEL / 32), blk, 0, stream>>>(
            W2l, wt, 4 * D_MODEL, D_MODEL);
        gemm_f16_kernel<false><<<dim3(D_MODEL / 128, ROWS / 128), blk, 0, stream>>>(
            act16, wt, B2l, x, x, nullptr, ROWS, D_MODEL, 4 * D_MODEL);
    }

    // h = LNf(x)  (f16)
    ln_kernel<<<ROWS, blk, 0, stream>>>(x, lnf_g, lnf_b, hh);

    // logits = h @ tok_emb^T
    dim3 lgrid((VOCAB + 127) / 128, ROWS / 128);
    if (use_tok16)
        gemm_f16_kernel<false><<<lgrid, blk, 0, stream>>>(
            hh, tok16, nullptr, nullptr, out, nullptr, ROWS, VOCAB, D_MODEL);
    else
        gemm_logits_f32b_kernel<<<lgrid, blk, 0, stream>>>(
            hh, tok, out, ROWS, VOCAB, D_MODEL);

    // loss (deterministic)
    float* loss = out + (size_t)ROWS * VOCAB;
    loss_row_kernel<<<ROWS, blk, 0, stream>>>(out, targets, rowls);
    loss_reduce_kernel<<<1, blk, 0, stream>>>(rowls, loss);
}
